// RNN_Discriminator_41102837023009
// MI455X (gfx1250) — matrix operand order repaired and checked by reading, not compiled, not run on hardware
//
#include <hip/hip_runtime.h>
#include <math.h>

// ---------------------------------------------------------------------------
// RNN discriminator, CDNA5 (gfx1250, wave32) WMMA implementation.
// B=64, T=512, D=512, K=4 fake seqs, V=10000.
//
// Pipeline:
//   1) prep_weights:   We/Wih/Whh f32 -> f16 in workspace (WMMA B operands,
//                      row-major [n][k] matches 32x16 B fragment layout).
//   2) compute_xi:     fused (embed gather + time linear) -> GEMM1+leaky ->
//                      GEMM2, xi stored f16 [NSEQ*B*T, D]. Fully parallel.
//   3) rnn_scan:       20 blocks (5 seq x 4 batch slices of 16). h state in
//                      LDS; per step: h@Whh^T via v_wmma_f32_16x16x32_f16,
//                      + xi + bhh, tanh, then Wo dot + sigmoid epilogue.
//   4) copy_masks:     mask passthrough outputs.
// ---------------------------------------------------------------------------

typedef __attribute__((ext_vector_type(16))) _Float16 v16h;
typedef __attribute__((ext_vector_type(8)))  _Float16 v8h;
typedef __attribute__((ext_vector_type(8)))  float    v8f;

#define BD   64
#define TD   512
#define DD   512
#define KKF  4
#define NSEQ 5
#define VS   520   // padded LDS row stride (halves); 520*2 bytes = 16B aligned

// ---- WMMA fragment loaders (wave32 layouts per CDNA5 ISA 7.12.2) ----------

// A: 16x32 f16 tile, rows in lanes 0..15 / 16..31, K split 0..15 vs
//    16..31 across the two half-waves.
__device__ __forceinline__ v16h afrag_lds(const _Float16* base, int lane, int kbase) {
  int row  = lane & 15;
  int koff = kbase + ((lane >> 4) << 4);
  v8h lo = *(const v8h*)(base + row * VS + koff);
  v8h hi = *(const v8h*)(base + row * VS + koff + 8);
  v16h a;
#pragma unroll
  for (int i = 0; i < 8; ++i) { a[i] = lo[i]; a[8 + i] = hi[i]; }
  return a;
}

// B: 32x16 f16 tile of B[k][n] = W[n][k]; lane n reads 16 contiguous k's of
//    row n of the row-major weight (lanes 16..31 take k+16). 32B per lane.
__device__ __forceinline__ v16h bfrag_w(const _Float16* W, int lane, int n0, int kbase) {
  int n  = n0 + (lane & 15);
  int kb = kbase + ((lane >> 4) << 4);
  return *(const v16h*)(W + (size_t)n * DD + kb);
}

// ---- 1) weight conversion -------------------------------------------------

__global__ void prep_weights(const float* We, const float* Wih, const float* Whh,
                             _Float16* wsh) {
  int total = 3 * DD * DD;
  for (int i = blockIdx.x * blockDim.x + threadIdx.x; i < total;
       i += gridDim.x * blockDim.x) {
    int m = i / (DD * DD), e = i - m * (DD * DD);
    const float* src = (m == 0) ? We : (m == 1) ? Wih : Whh;
    wsh[i] = (_Float16)src[e];
  }
}

// ---- 2) fused embed + GEMM1(leaky) + GEMM2 -> xi --------------------------

__global__ __launch_bounds__(256) void compute_xi_kernel(
    const int* __restrict__ real_marker, const float* __restrict__ real_time,
    const int* __restrict__ fake_marker, const float* __restrict__ fake_time,
    const float* __restrict__ emb, const float* __restrict__ be,
    const float* __restrict__ Wt, const float* __restrict__ bt,
    const float* __restrict__ bih, _Float16* __restrict__ ws) {
  __shared__ _Float16 vec_lds[16 * VS];
  __shared__ _Float16 x_lds[16 * VS];

  const _Float16* We_h  = ws;
  const _Float16* Wih_h = ws + (size_t)DD * DD;
  _Float16*       xi    = ws + (size_t)3 * DD * DD;

  const int tid  = threadIdx.x;
  const int lane = tid & 31;
  const int wave = tid >> 5;
  const int n0w  = wave * 64;
  const int mh   = lane >> 4;
  const int nl   = lane & 15;

  const int rowBase = blockIdx.x * 16;            // global row in [0, NSEQ*B*T)
  const int seq     = rowBase / (BD * TD);
  const int rem     = rowBase - seq * (BD * TD);  // b*T + t0, multiple of 16

  const int*   mk = (seq == 0) ? real_marker : fake_marker + (size_t)(seq - 1) * BD * TD;
  const float* tm = (seq == 0) ? real_time   : fake_time   + (size_t)(seq - 1) * BD * TD;

  // vec[r][d] = emb[marker] + BETA*(time*Wt[d] + bt[d]),  BETA = 1
  for (int e = tid; e < 16 * DD; e += 256) {
    int r = e >> 9;
    int d = e & (DD - 1);
    int row = rem + r;
    float v = emb[(size_t)mk[row] * DD + d] + tm[row] * Wt[d] + bt[d];
    vec_lds[r * VS + d] = (_Float16)v;
  }
  __syncthreads();

  // GEMM1: x = leaky(vec @ We^T + be)
  v8f acc[4];
#pragma unroll
  for (int i = 0; i < 4; ++i) acc[i] = (v8f){0.f,0.f,0.f,0.f,0.f,0.f,0.f,0.f};
  for (int kk = 0; kk < DD; kk += 32) {
    v16h a = afrag_lds(vec_lds, lane, kk);
#pragma unroll
    for (int nt = 0; nt < 4; ++nt) {
      v16h bf = bfrag_w(We_h, lane, n0w + nt * 16, kk);
      acc[nt] = __builtin_amdgcn_wmma_f32_16x16x32_f16(
          false, a, false, bf, (short)0, acc[nt], false, false);
    }
  }
#pragma unroll
  for (int nt = 0; nt < 4; ++nt) {
    int n = n0w + nt * 16 + nl;
    float bev = be[n];
#pragma unroll
    for (int r = 0; r < 8; ++r) {
      int m = r + 8 * mh;
      float v = acc[nt][r] + bev;
      v = (v >= 0.f) ? v : 0.01f * v;     // LeakyReLU(0.01)
      x_lds[m * VS + n] = (_Float16)v;
    }
  }
  __syncthreads();

  // GEMM2: xi = x @ Wih^T + bih  -> global f16
#pragma unroll
  for (int i = 0; i < 4; ++i) acc[i] = (v8f){0.f,0.f,0.f,0.f,0.f,0.f,0.f,0.f};
  for (int kk = 0; kk < DD; kk += 32) {
    v16h a = afrag_lds(x_lds, lane, kk);
#pragma unroll
    for (int nt = 0; nt < 4; ++nt) {
      v16h bf = bfrag_w(Wih_h, lane, n0w + nt * 16, kk);
      acc[nt] = __builtin_amdgcn_wmma_f32_16x16x32_f16(
          false, a, false, bf, (short)0, acc[nt], false, false);
    }
  }
#pragma unroll
  for (int nt = 0; nt < 4; ++nt) {
    int n = n0w + nt * 16 + nl;
    float biv = bih[n];
#pragma unroll
    for (int r = 0; r < 8; ++r) {
      int m = r + 8 * mh;
      xi[(size_t)(rowBase + m) * DD + n] = (_Float16)(acc[nt][r] + biv);
    }
  }
}

// ---- 3) sequential tanh-RNN scan + sigmoid head ---------------------------

__global__ __launch_bounds__(256) void rnn_scan_kernel(
    const float* __restrict__ real_mask, const float* __restrict__ fake_mask,
    const float* __restrict__ bhh, const float* __restrict__ Wo,
    const float* __restrict__ bo, const _Float16* __restrict__ ws,
    float* __restrict__ out) {
  __shared__ _Float16 h_lds[16 * VS];
  __shared__ float red[16];
  __shared__ float wo_lds[DD];

  const _Float16* Whh_h = ws + (size_t)2 * DD * DD;
  const _Float16* xi    = ws + (size_t)3 * DD * DD;

  const int tid  = threadIdx.x;
  const int lane = tid & 31;
  const int wave = tid >> 5;
  const int n0w  = wave * 64;
  const int mh   = lane >> 4;
  const int nl   = lane & 15;

  const int seq = blockIdx.x >> 2;          // 0..4
  const int bb  = (blockIdx.x & 3) * 16;    // batch slice base

  for (int e = tid; e < 16 * VS; e += 256) h_lds[e] = (_Float16)0.f;
  for (int e = tid; e < DD; e += 256) wo_lds[e] = Wo[e];
  if (tid < 16) red[tid] = 0.f;
  __syncthreads();

  const float* mask = (seq == 0) ? real_mask : fake_mask + (size_t)(seq - 1) * BD * TD;
  float* outR = (seq == 0) ? out : out + (size_t)2 * BD * TD + (size_t)(seq - 1) * BD * TD;

  float bhv[4], wov[4];
#pragma unroll
  for (int nt = 0; nt < 4; ++nt) {
    int n = n0w + nt * 16 + nl;
    bhv[nt] = bhh[n];
    wov[nt] = wo_lds[n];
  }
  const size_t xi_rowbase = ((size_t)seq * BD + bb) * TD;  // row of (seq,bb,t=0)

  for (int t = 0; t < TD; ++t) {
    // Prefetch next step's xi rows (global_prefetch_b8).
    if (t + 1 < TD) {
      const _Float16* nxt =
          xi + (xi_rowbase + (size_t)(tid >> 4) * TD + t + 1) * DD + (tid & 15) * 32;
      __builtin_prefetch(nxt, 0, 1);
    }

    // h_{t-1} @ Whh^T
    v8f acc[4];
#pragma unroll
    for (int i = 0; i < 4; ++i) acc[i] = (v8f){0.f,0.f,0.f,0.f,0.f,0.f,0.f,0.f};
    for (int kk = 0; kk < DD; kk += 32) {
      v16h a = afrag_lds(h_lds, lane, kk);
#pragma unroll
      for (int nt = 0; nt < 4; ++nt) {
        v16h bf = bfrag_w(Whh_h, lane, n0w + nt * 16, kk);
        acc[nt] = __builtin_amdgcn_wmma_f32_16x16x32_f16(
            false, a, false, bf, (short)0, acc[nt], false, false);
      }
    }
    __syncthreads();   // all waves done reading old h (and prev red consumed)

    // h_t = tanh(acc + xi + bhh); write back to LDS, keep for Wo dot
    if (tid < 16) red[tid] = 0.f;
    float ps[8];
#pragma unroll
    for (int r = 0; r < 8; ++r) ps[r] = 0.f;
#pragma unroll
    for (int nt = 0; nt < 4; ++nt) {
      int n = n0w + nt * 16 + nl;
#pragma unroll
      for (int r = 0; r < 8; ++r) {
        int m = r + 8 * mh;
        size_t row = xi_rowbase + (size_t)m * TD + t;
        float v = acc[nt][r] + (float)xi[row * DD + n] + bhv[nt];
        float h = tanhf(v);
        h_lds[m * VS + n] = (_Float16)h;
        ps[r] += h * wov[nt];
      }
    }
    __syncthreads();   // h updated, red zeroed

    // reduce Wo partials across the 16 lanes sharing each m
#pragma unroll
    for (int r = 0; r < 8; ++r) {
      float p = ps[r];
      p += __shfl_xor(p, 1);
      p += __shfl_xor(p, 2);
      p += __shfl_xor(p, 4);
      p += __shfl_xor(p, 8);
      if (nl == 0) atomicAdd(&red[r + 8 * mh], p);
    }
    __syncthreads();

    if (tid < 16) {
      int b = bb + tid;
      float rw = 1.f / (1.f + expf(-(red[tid] + bo[0])));
      outR[(size_t)b * TD + t] = rw * mask[(size_t)b * TD + t];
    }
  }
}

// ---- 4) mask passthrough --------------------------------------------------

__global__ void copy_masks(const float* __restrict__ real_mask,
                           const float* __restrict__ fake_mask,
                           float* __restrict__ out) {
  const int n1 = BD * TD, n2 = KKF * BD * TD;
  for (int i = blockIdx.x * blockDim.x + threadIdx.x; i < n1 + n2;
       i += gridDim.x * blockDim.x) {
    if (i < n1) out[n1 + i] = real_mask[i];
    else        out[2 * n1 + n2 + (i - n1)] = fake_mask[i - n1];
  }
}

// ---- launch ---------------------------------------------------------------

extern "C" void kernel_launch(void* const* d_in, const int* in_sizes, int n_in,
                              void* d_out, int out_size, void* d_ws, size_t ws_size,
                              hipStream_t stream) {
  (void)in_sizes; (void)n_in; (void)out_size; (void)ws_size;
  const int*   real_marker = (const int*)  d_in[0];
  const float* real_time   = (const float*)d_in[1];
  const float* real_mask   = (const float*)d_in[2];
  const int*   fake_marker = (const int*)  d_in[3];
  const float* fake_time   = (const float*)d_in[4];
  const float* fake_mask   = (const float*)d_in[5];
  const float* emb         = (const float*)d_in[6];
  const float* We          = (const float*)d_in[7];
  const float* be          = (const float*)d_in[8];
  const float* Wt          = (const float*)d_in[9];
  const float* bt          = (const float*)d_in[10];
  const float* Wih         = (const float*)d_in[11];
  const float* Whh         = (const float*)d_in[12];
  const float* bih         = (const float*)d_in[13];
  const float* bhh         = (const float*)d_in[14];
  const float* Wo          = (const float*)d_in[15];
  const float* bo          = (const float*)d_in[16];
  float*    out = (float*)d_out;
  _Float16* ws  = (_Float16*)d_ws;

  prep_weights<<<768, 256, 0, stream>>>(We, Wih, Whh, ws);
  compute_xi_kernel<<<NSEQ * BD * TD / 16, 256, 0, stream>>>(
      real_marker, real_time, fake_marker, fake_time, emb, be, Wt, bt, bih, ws);
  rnn_scan_kernel<<<NSEQ * 4, 256, 0, stream>>>(
      real_mask, fake_mask, bhh, Wo, bo, ws, out);
  copy_masks<<<512, 256, 0, stream>>>(real_mask, fake_mask, out);
}
